// BasicBlock_1w4a_LUT_13804024889751
// MI455X (gfx1250) — matrix operand order repaired and checked by reading, not compiled, not run on hardware
//
#include <hip/hip_runtime.h>

typedef __attribute__((ext_vector_type(16))) _Float16 v16h;
typedef __attribute__((ext_vector_type(8)))  _Float16 v8h;
typedef __attribute__((ext_vector_type(8)))  float    v8f;

#define CIN   64
#define COUT  64
#define HH    112
#define WW    112
#define KCH   18      // 576 / 32 K-chunks, ordered (tap, ci)
#define NT    4       // 4 N-tiles of 16 channels

union AV { v16h v; v8h h[2]; };
union PK { _Float16 h[2]; unsigned u; };

// ---------------------------------------------------------------------------
// Kernel 1: binarize weights (mean-center, unbiased std, pow2 scale, sign)
// and pre-pack into this kernel's WMMA B-fragment layout (contiguous k):
//   packedB[ntile][chunk][lane][half]  (f16)
//   lane  = (K_within>=16 ? 16 : 0) + (ch & 15)
//   half  = K_within & 15,  K = (kh*3+kw)*64 + ci
// ---------------------------------------------------------------------------
__global__ __launch_bounds__(64) void prep_weights(const float* __restrict__ w,
                                                   _Float16* __restrict__ packedB) {
  __shared__ float red[64];
  const int c = blockIdx.x;          // output channel
  const int t = threadIdx.x;
  const float* wc = w + c * 576;

  float vals[9];
  float s = 0.f;
  for (int j = 0; j < 9; ++j) { vals[j] = wc[t + 64 * j]; s += vals[j]; }
  red[t] = s; __syncthreads();
  for (int off = 32; off > 0; off >>= 1) { if (t < off) red[t] += red[t + off]; __syncthreads(); }
  const float mean = red[0] / 576.f;
  __syncthreads();

  float ssq = 0.f, sab = 0.f;
  for (int j = 0; j < 9; ++j) { float d = vals[j] - mean; ssq += d * d; sab += fabsf(d); }
  red[t] = ssq; __syncthreads();
  for (int off = 32; off > 0; off >>= 1) { if (t < off) red[t] += red[t + off]; __syncthreads(); }
  const float var = red[0] / 575.f;   // ddof=1
  __syncthreads();
  red[t] = sab; __syncthreads();
  for (int off = 32; off > 0; off >>= 1) { if (t < off) red[t] += red[t + off]; __syncthreads(); }
  const float sdev = sqrtf(var);
  const float mean_abs = red[0] / (576.f * sdev);
  const float sw = exp2f(rintf(log2f(mean_abs)));   // pow2 scale, round-half-even

  const int ntile = c >> 4;
  for (int j = 0; j < 9; ++j) {
    const int idx = t + 64 * j;           // = ci*9 + tap  (OIHW)
    const int ci = idx / 9, tap = idx % 9;
    const float d = vals[j] - mean;
    const float bv = (d > 0.f) ? sw : ((d < 0.f) ? -sw : 0.f);  // sign()*sw
    const int K = tap * 64 + ci;
    const int chunk = K >> 5, within = K & 31;
    const int lane = ((within >> 4) << 4) + (c & 15);
    const int half = within & 15;
    packedB[(((ntile * KCH + chunk) * 32) + lane) * 16 + half] = (_Float16)bv;
  }
}

// ---------------------------------------------------------------------------
// Kernel 2: implicit-GEMM conv via v_wmma_f32_16x16x32_f16 + LUT bucketize.
// Workgroup = 224 threads (7 wave32), computes 2 output rows x 112 w x 64 ch.
// LDS stage: 4 input rows (padded to 114 w) x 64 ci as f16, layout [r][w][ci].
// ---------------------------------------------------------------------------
__global__ __launch_bounds__(224) void conv_lut(const float* __restrict__ x,
                                                const _Float16* __restrict__ packedB,
                                                const float* __restrict__ lut,
                                                float* __restrict__ out) {
  __shared__ __align__(16) _Float16 xs[4 * 114 * 64];   // 58,368 B
  const int b  = blockIdx.x / 56;
  const int h0 = (blockIdx.x % 56) * 2;
  const int tid = threadIdx.x;

  unsigned* xsu = (unsigned*)xs;
  // zero the w-pad columns (wp = 0 and 113)
  for (int j = tid; j < 256; j += 224) {
    const int r = j >> 6, wp = ((j >> 5) & 1) ? 113 : 0, cip = j & 31;
    xsu[((r * 114 + wp) * 64 + cip * 2) >> 1] = 0u;
  }
  // stage rows h0-1 .. h0+2 : coalesced f32 loads, pack 2 ci into one dword
  for (int i = 0; i < 64; ++i) {
    const int g = tid + i * 224;
    const int wcol = g % 112, grp = g / 112;
    const int r = grp >> 5, ci = (grp & 31) * 2;
    const int hin = h0 - 1 + r;
    float a0 = 0.f, a1 = 0.f;
    if (hin >= 0 && hin < HH) {
      const long base = ((long)(b * CIN + ci) * HH + hin) * WW + wcol;
      a0 = x[base];
      a1 = x[base + (long)HH * WW];
    }
    PK p; p.h[0] = (_Float16)a0; p.h[1] = (_Float16)a1;
    xsu[((r * 114 + (wcol + 1)) * 64 + ci) >> 1] = p.u;
  }
  __syncthreads();

  const int lane  = tid & 31;
  const int wid   = tid >> 5;          // 0..6
  const int lhalf = lane >> 4;         // 0/1 lane group
  const int l16   = lane & 15;

  // each wave owns m-tiles (wid*2, wid*2+1); 14 m-tiles = 2 rows x 7 w-tiles
  int rl_[2], wbase_[2];
  for (int s = 0; s < 2; ++s) { const int mt = wid * 2 + s; rl_[s] = mt / 7; wbase_[s] = (mt % 7) * 16; }

  const v8f vzero = {0.f, 0.f, 0.f, 0.f, 0.f, 0.f, 0.f, 0.f};
  v8f acc[2][4];
  for (int s = 0; s < 2; ++s)
    for (int n = 0; n < NT; ++n) acc[s][n] = vzero;

  const v16h* bp = (const v16h*)packedB;

  for (int c = 0; c < KCH; ++c) {
    const int tap = c >> 1;
    const int kh = tap / 3, kw = tap % 3;
    // A halves 0..15 -> ci cib..cib+15 (contiguous)
    const int cib = ((c & 1) << 5) + (lhalf << 4);
    AV a[2];
#pragma unroll
    for (int s = 0; s < 2; ++s) {
      const int rin = rl_[s] + kh;
      const int wp  = wbase_[s] + l16 + kw;
      const int off = (rin * 114 + wp) * 64 + cib;
      a[s].h[0] = *(const v8h*)(xs + off);        // ds_load_b128
      a[s].h[1] = *(const v8h*)(xs + off + 8);    // ds_load_b128 (+16B)
    }
#pragma unroll
    for (int n = 0; n < NT; ++n) {
      const v16h bf = bp[(n * KCH + c) * 32 + lane];   // contiguous 32B
#pragma unroll
      for (int s = 0; s < 2; ++s)
        acc[s][n] = __builtin_amdgcn_wmma_f32_16x16x32_f16(
            false, a[s].v, false, bf, (short)0, acc[s][n], false, false);
    }
  }

  __syncthreads();                       // xs no longer needed; reuse as out staging
  float* obuf = (float*)xs + wid * 256;  // 16x16 f32 per wave

#pragma unroll
  for (int n = 0; n < NT; ++n) {
    const int ch = n * 16 + l16;         // this lane's channel in C/D layout
    const float t0 = lut[ch * 7 + 0], t1 = lut[ch * 7 + 1], t2 = lut[ch * 7 + 2];
    const float t3 = lut[ch * 7 + 3], t4 = lut[ch * 7 + 4], t5 = lut[ch * 7 + 5];
    const float t6 = lut[ch * 7 + 6];
    for (int s = 0; s < 2; ++s) {
#pragma unroll
      for (int i = 0; i < 8; ++i) {
        const float y = acc[s][n][i];
        float r = y;  // sequential where-order, faithful to reference
        r = (y <= t0)            ? 0.f : r;
        r = (y >  t0 && y <  t1) ? 1.f : r;
        r = (y >= t1 && y <= t2) ? 2.f : r;
        r = (y >  t2 && y <  t3) ? 3.f : r;
        r = (y >= t3 && y <= t4) ? 4.f : r;
        r = (y >  t4 && y <  t5) ? 5.f : r;
        r = (y >= t5 && y <= t6) ? 6.f : r;
        r = (y >  t6)            ? 7.f : r;
        obuf[l16 * 16 + (lhalf << 3) + i] = r;    // [ch][pixel] transpose buffer
      }
      const int hrow = h0 + rl_[s];
      const int wc0  = wbase_[s];
#pragma unroll
      for (int k = 0; k < 8; ++k) {               // coalesced: 2ch x 16w per store
        const int c2 = 2 * k + lhalf;
        const float v = obuf[c2 * 16 + l16];
        const long oaddr = (((long)(b * COUT + n * 16 + c2)) * HH + hrow) * WW + wc0 + l16;
        out[oaddr] = v;
      }
    }
  }
}

extern "C" void kernel_launch(void* const* d_in, const int* in_sizes, int n_in,
                              void* d_out, int out_size, void* d_ws, size_t ws_size,
                              hipStream_t stream) {
  (void)in_sizes; (void)n_in; (void)out_size; (void)ws_size;
  const float* x   = (const float*)d_in[0];
  const float* w   = (const float*)d_in[1];
  const float* lut = (const float*)d_in[2];
  float* out = (float*)d_out;
  _Float16* packedB = (_Float16*)d_ws;   // 4*18*32*16 halves = 73,728 B

  prep_weights<<<64, 64, 0, stream>>>(w, packedB);
  conv_lut<<<32 * 56, 224, 0, stream>>>(x, packedB, lut, out);
}
